// MambaLayer_22376779612654
// MI455X (gfx1250) — matrix operand order repaired and checked by reading, not compiled, not run on hardware
//
#include <hip/hip_runtime.h>
#include <hip/hip_bf16.h>
#include <math.h>

// ---------------- problem constants ----------------
#define DIMC    192
#define DINNER  384
#define DPROJ   768          // 2*DINNER
#define DTRANK  12
#define NPROJ   14           // DTRANK + 2*D_STATE
#define BATCH   16
#define LSEQ    9216         // 96*96
#define NTOK    (BATCH*LSEQ) // 147456
#define CHUNK   256
#define NCHUNK  (LSEQ/CHUNK) // 36
#define EPSLN   1e-5f

typedef __attribute__((ext_vector_type(16))) __bf16 v16bf;
typedef __attribute__((ext_vector_type(8)))  __bf16 v8bf;
typedef __attribute__((ext_vector_type(8)))  float  v8f;
typedef __attribute__((ext_vector_type(4)))  unsigned int v4u;
typedef __attribute__((ext_vector_type(8)))  int v8i;
typedef __attribute__((ext_vector_type(4)))  int v4i;

union Frag16 { v16bf v; v8bf h[2]; };

__device__ __forceinline__ __bf16 tobf(float f) { return (__bf16)f; }
__device__ __forceinline__ float  fromf(__bf16 b) { return (float)b; }
__device__ __forceinline__ float  siluf(float x) { return x / (1.f + __expf(-x)); }
__device__ __forceinline__ float  softplusf(float x) { return (x > 20.f) ? x : log1pf(__expf(x)); }

// ---------------------------------------------------------------------
// TDM: contiguous global->LDS copy via Tensor Data Mover.
//   bytes must be a multiple of 8 and <= 65535*8.  D# per ISA ch.8:
//   group0 = {count=1 | lds_addr | global_addr | type=2}
//   group1 = {mask=0, data_size=8B, tensor_dim0=n8, tensor_dim1=1,
//             tile_dim0=n8, tile_dim1=1, dim0_stride=n8}
//   Issue from a single wave; EXEC is ignored by tensor ops.
//   (6-arg clang-23 form: extra zero-filled int32x8 group before cpol.)
// ---------------------------------------------------------------------
__device__ __forceinline__ void tdm_load_to_lds(void* lds_dst, const void* gsrc,
                                                unsigned bytes) {
    const unsigned n8 = bytes >> 3;
    const unsigned long long ga = (unsigned long long)gsrc;
    const unsigned ldsoff = (unsigned)(unsigned long long)lds_dst; // low 32b = LDS offset
    v4u g0;
    g0.x = 1u;                                            // count=1 (valid user D#)
    g0.y = ldsoff;                                        // lds_addr
    g0.z = (unsigned)ga;                                  // global_addr[31:0]
    g0.w = (unsigned)((ga >> 32) & 0x01FFFFFFull) | 0x80000000u; // [56:32] | type=2
    v8i g1;
    g1[0] = (int)(3u << 16);                              // data_size = 8 bytes
    g1[1] = (int)((n8 & 0xFFFFu) << 16);                  // tensor_dim0[15:0]
    g1[2] = (int)(((n8 >> 16) & 0xFFFFu) | (1u << 16));   // tensor_dim0[31:16], tensor_dim1=1
    g1[3] = (int)((n8 & 0xFFFFu) << 16);                  // tile_dim0 = n8
    g1[4] = 1;                                            // tile_dim1 = 1
    g1[5] = (int)n8;                                      // tensor_dim0_stride[31:0]
    g1[6] = 0;
    g1[7] = 0;
    v4i g2 = {0, 0, 0, 0};
    v4i g3 = {0, 0, 0, 0};
    v8i g4 = {0, 0, 0, 0, 0, 0, 0, 0};
    __builtin_amdgcn_tensor_load_to_lds(g0, g1, g2, g3, g4, 0);
}

// =====================================================================
// K0: pack W_in / W_out into WMMA-B fragment layout (bf16), A = -exp(A_log)
// =====================================================================
__global__ void k_pack(const float* __restrict__ W_in, const float* __restrict__ W_out,
                       const float* __restrict__ A_log,
                       __bf16* __restrict__ pWin, __bf16* __restrict__ pWout,
                       float* __restrict__ Aneg) {
    const int tid = threadIdx.x;
    for (int t = tid; t < 288 * 512; t += blockDim.x) {
        int tile = t >> 9, r = t & 511, lane = r >> 4, j = r & 15;
        int kt = tile % 6, nt = tile / 6;
        int k = kt * 32 + (lane >> 4) * 8 + (j & 7) + (j >> 3) * 16;
        int n = nt * 16 + (lane & 15);
        pWin[t] = tobf(W_in[k * DPROJ + n]);
    }
    for (int t = tid; t < 144 * 512; t += blockDim.x) {
        int tile = t >> 9, r = t & 511, lane = r >> 4, j = r & 15;
        int kt = tile % 12, nt = tile / 12;
        int k = kt * 32 + (lane >> 4) * 8 + (j & 7) + (j >> 3) * 16;
        int n = nt * 16 + (lane & 15);
        pWout[t] = tobf(W_out[k * DIMC + n]);
    }
    for (int c = tid; c < DINNER; c += blockDim.x)
        Aneg[c] = -__expf(A_log[c]);
}

// =====================================================================
// K1: LayerNorm, one wave per token
// =====================================================================
__global__ __launch_bounds__(256) void k_ln(const float* __restrict__ x,
                                            const float* __restrict__ w,
                                            const float* __restrict__ b,
                                            __bf16* __restrict__ xn) {
    const int lane = threadIdx.x & 31;
    const int tok  = blockIdx.x * 8 + (threadIdx.x >> 5);
    const size_t base = (size_t)tok * DIMC;
    float v[6], s = 0.f, sq = 0.f;
#pragma unroll
    for (int j = 0; j < 6; ++j) {
        v[j] = x[base + lane + 32 * j];
        s += v[j]; sq += v[j] * v[j];
    }
#pragma unroll
    for (int m = 16; m >= 1; m >>= 1) { s += __shfl_xor(s, m); sq += __shfl_xor(sq, m); }
    const float mean = s * (1.f / DIMC);
    const float var  = sq * (1.f / DIMC) - mean * mean;
    const float rstd = rsqrtf(var + EPSLN);
#pragma unroll
    for (int j = 0; j < 6; ++j) {
        int c = lane + 32 * j;
        xn[base + c] = tobf((v[j] - mean) * rstd * w[c] + b[c]);
    }
}

// =====================================================================
// K2: GEMM1  xz = xn(bf16) @ W_in  ->  xb / z  (bf16), W_in halves via TDM
// =====================================================================
__global__ __launch_bounds__(256) void k_gemm_in(const __bf16* __restrict__ xn,
                                                 const __bf16* __restrict__ pWin,
                                                 __bf16* __restrict__ xb,
                                                 __bf16* __restrict__ z) {
    extern __shared__ __align__(16) unsigned char smem[];
    __bf16* lds = (__bf16*)smem;                 // 73728 bf16 = 147456 B per half
    const int tid = threadIdx.x;
    const int lane = tid & 31, wave = tid >> 5;
    const int tokbase = (blockIdx.x * 8 + wave) * 16;
    const int m = lane & 15, khalf = lane >> 4;

    Frag16 a[6];
#pragma unroll
    for (int kt = 0; kt < 6; ++kt) {
        size_t e = (size_t)(tokbase + m) * DIMC + kt * 32 + khalf * 8;
        a[kt].h[0] = *(const v8bf*)(xn + e);
        a[kt].h[1] = *(const v8bf*)(xn + e + 16);
    }

    for (int hh = 0; hh < 2; ++hh) {
        __syncthreads();                          // previous half fully consumed
        if (wave == 0) {
            tdm_load_to_lds(lds, pWin + (size_t)hh * 73728, 147456u);
            __builtin_amdgcn_s_wait_tensorcnt(0);
        }
        __syncthreads();
#pragma unroll 4
        for (int ntl = 0; ntl < 24; ++ntl) {
            v8f acc = {};
#pragma unroll
            for (int kt = 0; kt < 6; ++kt) {
                Frag16 bf;
                const __bf16* bp = lds + ((ntl * 6 + kt) * 32 + lane) * 16;
                bf.h[0] = *(const v8bf*)(bp);
                bf.h[1] = *(const v8bf*)(bp + 8);
                acc = __builtin_amdgcn_wmma_f32_16x16x32_bf16(
                    false, a[kt].v, false, bf.v, (short)0, acc, false, false);
            }
            const int nt = hh * 24 + ntl;
            const int colg = nt * 16 + (lane & 15);
            __bf16* dstbuf = (nt < 24) ? xb : z;
            const int col = (nt < 24) ? colg : (colg - DINNER);
#pragma unroll
            for (int i = 0; i < 8; ++i) {
                int row = tokbase + i + khalf * 8;
                dstbuf[(size_t)row * DINNER + col] = tobf(acc[i]);
            }
        }
    }
}

// =====================================================================
// K3: fused conv3+SiLU -> u ; proj 384->14 (wave reduce) ; dt ; dA,dBu,C
// =====================================================================
__global__ __launch_bounds__(256) void k_conv_proj(const __bf16* __restrict__ xb,
                                                   const float* __restrict__ conv_w,
                                                   const float* __restrict__ conv_b,
                                                   const float* __restrict__ W_xproj,
                                                   const float* __restrict__ W_dt,
                                                   const float* __restrict__ b_dt,
                                                   const float* __restrict__ Aneg,
                                                   __bf16* __restrict__ u_out,
                                                   __bf16* __restrict__ dA_out,
                                                   __bf16* __restrict__ dBu_out,
                                                   float* __restrict__ C_out) {
    __shared__ float sWx[DINNER * NPROJ];   // 21504 B
    __shared__ float sWdt[DTRANK * DINNER]; // 18432 B
    __shared__ float sBdt[DINNER];          //  1536 B
    const int tid = threadIdx.x;
    if (tid < 32) {   // wave 0 stages all weights through the TDM
        tdm_load_to_lds(sWx,  W_xproj, DINNER * NPROJ * 4u);
        tdm_load_to_lds(sWdt, W_dt,    DTRANK * DINNER * 4u);
        tdm_load_to_lds(sBdt, b_dt,    DINNER * 4u);
        __builtin_amdgcn_s_wait_tensorcnt(0);
    }
    __syncthreads();

    const int lane = tid & 31;
    const int tok  = blockIdx.x * 8 + (tid >> 5);
    const int ll   = tok % LSEQ;
    const size_t rbase = (size_t)tok * DINNER;

    float uv[12];
    float p[NPROJ];
#pragma unroll
    for (int o = 0; o < NPROJ; ++o) p[o] = 0.f;

#pragma unroll
    for (int j = 0; j < 12; ++j) {
        const int c = lane + 32 * j;
        float x0 = (ll >= 2) ? fromf(xb[rbase - 2 * DINNER + c]) : 0.f;
        float x1 = (ll >= 1) ? fromf(xb[rbase - 1 * DINNER + c]) : 0.f;
        float x2 = fromf(xb[rbase + c]);
        float acc = conv_w[c * 3 + 0] * x0 + conv_w[c * 3 + 1] * x1 +
                    conv_w[c * 3 + 2] * x2 + conv_b[c];
        float u = siluf(acc);
        uv[j] = u;
        u_out[rbase + c] = tobf(u);
#pragma unroll
        for (int o = 0; o < NPROJ; ++o) p[o] = fmaf(u, sWx[c * NPROJ + o], p[o]);
    }
#pragma unroll
    for (int o = 0; o < NPROJ; ++o) {
        float v = p[o];
#pragma unroll
        for (int m = 16; m >= 1; m >>= 1) v += __shfl_xor(v, m);
        p[o] = v;
    }
    const float Bs = p[DTRANK], Cs = p[DTRANK + 1];
#pragma unroll
    for (int j = 0; j < 12; ++j) {
        const int c = lane + 32 * j;
        float d = sBdt[c];
#pragma unroll
        for (int r = 0; r < DTRANK; ++r) d = fmaf(p[r], sWdt[r * DINNER + c], d);
        const float dt = softplusf(d);
        dA_out[rbase + c]  = tobf(__expf(dt * Aneg[c]));
        dBu_out[rbase + c] = tobf(dt * Bs * uv[j]);
    }
    if (lane == 0) C_out[tok] = Cs;
}

// =====================================================================
// K4: scan stage A — per (batch, chunk): (prod dA, h_end) per channel
// =====================================================================
__global__ __launch_bounds__(384) void k_scanA(const __bf16* __restrict__ dA,
                                               const __bf16* __restrict__ dBu,
                                               float* __restrict__ cPA,
                                               float* __restrict__ cH) {
    const int b = blockIdx.x / NCHUNK, ch = blockIdx.x % NCHUNK, c = threadIdx.x;
    size_t idx = ((size_t)(b * LSEQ + ch * CHUNK)) * DINNER + c;
    float h = 0.f, pr = 1.f;
    for (int i = 0; i < CHUNK; ++i, idx += DINNER) {
        if ((i & 15) == 0) {  // stream-ahead prefetch, 32 rows
            __builtin_prefetch((const void*)(dA  + idx + 32 * DINNER), 0, 1);
            __builtin_prefetch((const void*)(dBu + idx + 32 * DINNER), 0, 1);
        }
        float a = fromf(dA[idx]);
        h = fmaf(a, h, fromf(dBu[idx]));
        pr *= a;
    }
    const size_t o = (size_t)(b * NCHUNK + ch) * DINNER + c;
    cPA[o] = pr; cH[o] = h;
}

// =====================================================================
// K5: scan stage B — prefix over the 36 chunks, per (batch, channel)
// =====================================================================
__global__ __launch_bounds__(384) void k_scanB(const float* __restrict__ cPA,
                                               const float* __restrict__ cH,
                                               float* __restrict__ cPref) {
    const int b = blockIdx.x, c = threadIdx.x;
    float h = 0.f;
    for (int ch = 0; ch < NCHUNK; ++ch) {
        const size_t o = (size_t)(b * NCHUNK + ch) * DINNER + c;
        cPref[o] = h;
        h = fmaf(cPA[o], h, cH[o]);
    }
}

// =====================================================================
// K6: scan stage C — rescan with correct h0, fuse +u*D and *silu(z) -> yg
// =====================================================================
__global__ __launch_bounds__(384) void k_scanC(const __bf16* __restrict__ dA,
                                               const __bf16* __restrict__ dBu,
                                               const __bf16* __restrict__ u,
                                               const __bf16* __restrict__ z,
                                               const float* __restrict__ Cssm,
                                               const float* __restrict__ Dvec,
                                               const float* __restrict__ cPref,
                                               __bf16* __restrict__ yg) {
    const int b = blockIdx.x / NCHUNK, ch = blockIdx.x % NCHUNK, c = threadIdx.x;
    const int tok0 = b * LSEQ + ch * CHUNK;
    float h = cPref[(size_t)(b * NCHUNK + ch) * DINNER + c];
    const float dc = Dvec[c];
    size_t idx = (size_t)tok0 * DINNER + c;
    for (int i = 0; i < CHUNK; ++i, idx += DINNER) {
        if ((i & 15) == 0) {
            __builtin_prefetch((const void*)(dA  + idx + 32 * DINNER), 0, 1);
            __builtin_prefetch((const void*)(dBu + idx + 32 * DINNER), 0, 1);
            __builtin_prefetch((const void*)(u   + idx + 32 * DINNER), 0, 1);
            __builtin_prefetch((const void*)(z   + idx + 32 * DINNER), 0, 1);
        }
        float a = fromf(dA[idx]);
        h = fmaf(a, h, fromf(dBu[idx]));
        float y = fmaf(h, Cssm[tok0 + i], fromf(u[idx]) * dc);
        float zv = fromf(z[idx]);
        yg[idx] = tobf(y * siluf(zv));
    }
}

// =====================================================================
// K7: GEMM2  out = yg(bf16) @ W_out + x ; W_out in LDS via TDM
// =====================================================================
__global__ __launch_bounds__(256) void k_gemm_out(const __bf16* __restrict__ yg,
                                                  const __bf16* __restrict__ pWout,
                                                  const float* __restrict__ x,
                                                  float* __restrict__ out) {
    extern __shared__ __align__(16) unsigned char smem[];
    __bf16* lds = (__bf16*)smem;                 // 73728 bf16 = 147456 B
    const int tid = threadIdx.x;
    const int lane = tid & 31, wave = tid >> 5;
    if (wave == 0) {
        tdm_load_to_lds(lds, pWout, 147456u);
        __builtin_amdgcn_s_wait_tensorcnt(0);
    }
    __syncthreads();

    const int tokbase = (blockIdx.x * 8 + wave) * 16;
    const int m = lane & 15, khalf = lane >> 4;

    v8f acc[12];
#pragma unroll
    for (int nt = 0; nt < 12; ++nt) acc[nt] = (v8f){};

#pragma unroll 2
    for (int kt = 0; kt < 12; ++kt) {
        Frag16 a;
        size_t e = (size_t)(tokbase + m) * DINNER + kt * 32 + khalf * 8;
        a.h[0] = *(const v8bf*)(yg + e);
        a.h[1] = *(const v8bf*)(yg + e + 16);
#pragma unroll
        for (int nt = 0; nt < 12; ++nt) {
            Frag16 bf;
            const __bf16* bp = lds + ((nt * 12 + kt) * 32 + lane) * 16;
            bf.h[0] = *(const v8bf*)(bp);
            bf.h[1] = *(const v8bf*)(bp + 8);
            acc[nt] = __builtin_amdgcn_wmma_f32_16x16x32_bf16(
                false, a.v, false, bf.v, (short)0, acc[nt], false, false);
        }
    }
#pragma unroll
    for (int nt = 0; nt < 12; ++nt) {
        const int col = nt * 16 + (lane & 15);
#pragma unroll
        for (int i = 0; i < 8; ++i) {
            int row = tokbase + i + khalf * 8;
            size_t idx = (size_t)row * DIMC + col;
            out[idx] = acc[nt][i] + x[idx];
        }
    }
}

// =====================================================================
extern "C" void kernel_launch(void* const* d_in, const int* in_sizes, int n_in,
                              void* d_out, int out_size, void* d_ws, size_t ws_size,
                              hipStream_t stream) {
    const float* x      = (const float*)d_in[0];
    const float* ln_w   = (const float*)d_in[1];
    const float* ln_b   = (const float*)d_in[2];
    const float* W_in   = (const float*)d_in[3];
    const float* conv_w = (const float*)d_in[4];
    const float* conv_b = (const float*)d_in[5];
    const float* W_xprj = (const float*)d_in[6];
    const float* W_dt   = (const float*)d_in[7];
    const float* b_dt   = (const float*)d_in[8];
    const float* A_log  = (const float*)d_in[9];
    const float* Dvec   = (const float*)d_in[10];
    const float* W_out  = (const float*)d_in[11];
    float* out = (float*)d_out;

    size_t off = 0;
    auto carve = [&](size_t bytes) -> void* {
        void* p = (char*)d_ws + off;
        off += (bytes + 255) & ~(size_t)255;
        return p;
    };
    __bf16* xn    = (__bf16*)carve((size_t)NTOK * DIMC * 2);
    __bf16* pWin  = (__bf16*)carve(288 * 512 * 2);
    __bf16* pWout = (__bf16*)carve(144 * 512 * 2);
    float*  Aneg  = (float*) carve(DINNER * 4);
    __bf16* xb    = (__bf16*)carve((size_t)NTOK * DINNER * 2); // reused as yg
    __bf16* zbuf  = (__bf16*)carve((size_t)NTOK * DINNER * 2);
    __bf16* ubuf  = (__bf16*)carve((size_t)NTOK * DINNER * 2);
    __bf16* dA    = (__bf16*)carve((size_t)NTOK * DINNER * 2);
    __bf16* dBu   = (__bf16*)carve((size_t)NTOK * DINNER * 2);
    float*  Cssm  = (float*) carve((size_t)NTOK * 4);
    float*  cPA   = (float*) carve((size_t)BATCH * NCHUNK * DINNER * 4);
    float*  cH    = (float*) carve((size_t)BATCH * NCHUNK * DINNER * 4);
    float*  cPref = (float*) carve((size_t)BATCH * NCHUNK * DINNER * 4);
    __bf16* yg    = xb; // reuse

    k_pack<<<1, 256, 0, stream>>>(W_in, W_out, A_log, pWin, pWout, Aneg);
    k_ln<<<NTOK / 8, 256, 0, stream>>>(x, ln_w, ln_b, xn);
    k_gemm_in<<<NTOK / 128, 256, 147456, stream>>>(xn, pWin, xb, zbuf);
    k_conv_proj<<<NTOK / 8, 256, 0, stream>>>(xb, conv_w, conv_b, W_xprj, W_dt,
                                              b_dt, Aneg, ubuf, dA, dBu, Cssm);
    k_scanA<<<BATCH * NCHUNK, 384, 0, stream>>>(dA, dBu, cPA, cH);
    k_scanB<<<BATCH, 384, 0, stream>>>(cPA, cH, cPref);
    k_scanC<<<BATCH * NCHUNK, 384, 0, stream>>>(dA, dBu, ubuf, zbuf, Cssm, Dvec,
                                                cPref, yg);
    k_gemm_out<<<NTOK / 128, 256, 147456, stream>>>(yg, pWout, x, out);
}
